// Conv_67954972557517
// MI455X (gfx1250) — matrix operand order repaired and checked by reading, not compiled, not run on hardware
//
#include <hip/hip_runtime.h>
#include <hip/hip_bf16.h>
#include <climits>

typedef __attribute__((ext_vector_type(16))) _Float16       v16h;
typedef __attribute__((ext_vector_type(16))) unsigned short v16us;
typedef __attribute__((ext_vector_type(8)))  float          v8f;

// ---- problem constants (from reference) ----
#define T_    15
#define CIN   40
#define H_    160
#define W_    300
#define COUT  50
#define KH    4
#define KW    6
#define OH    157          // 160-4+1
#define OW    295          // 300-6+1
#define KDIM  960          // CIN*KH*KW
#define KSTEPS 30          // KDIM / 32
#define MTILES 4           // 64 output channels (50 valid, rest padded)
#define TCOL  128          // ow tile per block
#define NROWS (CIN*KH)     // 160 (cin,kh) rows in the raw LDS tile
#define ROWPAD 136         // 133 needed (128+6-1), padded
#define PH    52           // floor(157/3)
#define PW    98           // floor(295/3)
#define THRESH 15.0f

// ---- workspace layout ----
#define MINIDX_OFF 123904  // byte offset of the int cell (Apack = 122880 B)

// -----------------------------------------------------------------------------
// Pack weights into this file's A-fragment k order (contiguous, same as B; 16-bit A 16x32, wave32):
//   lanes 0-15 : row M = lane,    K = {0..15}
//   lanes 16-31: row M = lane-16, K = {16..31}
//   element e:  kLocal = e + (lane>=16 ? 16:0)
// -----------------------------------------------------------------------------
__global__ void pack_A(const float* __restrict__ Wt, _Float16* __restrict__ Apack) {
    const int bx    = blockIdx.x;          // mtile*KSTEPS + kstep
    const int mtile = bx / KSTEPS;
    const int kstep = bx % KSTEPS;
    const int lane  = threadIdx.x;         // 0..31
    const int co    = mtile * 16 + (lane & 15);
    const int kadd  = (lane >= 16) ? 16 : 0;

    v16h tmp;
#pragma unroll
    for (int e = 0; e < 16; ++e) {
        const int v  = e >> 1, hb = e & 1;
        const int kl = e + kadd; (void)v; (void)hb;
        const int k  = kstep * 32 + kl;                 // <= 959, always in range
        const float val = (co < COUT) ? Wt[co * KDIM + k] : 0.0f;
        tmp[e] = (_Float16)val;
    }
    *reinterpret_cast<v16h*>(Apack + ((size_t)bx * 32 + lane) * 16) = tmp;
}

// -----------------------------------------------------------------------------
// Implicit-GEMM conv via v_wmma_f32_16x16x32_f16, LDS-centric hot loop with a
// double-buffered fragment slab (one s_barrier per k-step).
// Block = 256 threads (8 waves). Tile: 64 channels x 128 ow, one (t, oh).
// -----------------------------------------------------------------------------
__global__ __launch_bounds__(256)
void conv_wmma(const float* __restrict__ x, const _Float16* __restrict__ Apack,
               float* __restrict__ pots) {
    __shared__ _Float16 xs[NROWS * ROWPAD];            // 43,520 B raw tile (f16)
    __shared__ v16h     Bs[2][256];                    // 16,384 B ping-pong slab
    __shared__ __align__(32) unsigned short offB[KDIM];// 1,920 B im2col offsets

    const int tid  = threadIdx.x;
    const int wave = tid >> 5;
    const int lane = tid & 31;
    const int ow0  = blockIdx.x * TCOL;
    const int oh   = blockIdx.y;
    const int t    = blockIdx.z;

    // ---- stage raw x tile into LDS as f16 (once per block, coalesced) ----
    const float* xrow = x + ((size_t)t * CIN) * H_ * W_ + (size_t)oh * W_;
    for (int idx = tid; idx < NROWS * ROWPAD; idx += 256) {
        const int row = idx / ROWPAD;                  // cin*4 + kh
        const int col = idx - row * ROWPAD;
        const int cin = row >> 2;
        const int kh  = row & 3;
        const int gw  = ow0 + col;
        float v = 0.0f;
        if (col < TCOL + KW - 1 && gw < W_)
            v = xrow[((size_t)cin * H_ + kh) * W_ + gw];
        xs[idx] = (_Float16)v;                         // zero-pad tail columns
    }
    // ---- im2col offset table (half-index at column 0), once per block ----
    for (int k = tid; k < KDIM; k += 256) {
        const int cin = k / 24;
        const int r   = k - cin * 24;
        const int kh  = r / 6;
        const int kw  = r - kh * 6;
        offB[k] = (unsigned short)((cin * KH + kh) * ROWPAD + kw);
    }

    // this thread's B-fill slot: subtile = wave, column = lane&15
    const int owLoc = wave * 16 + (lane & 15);         // column within 128-tile
    const int kadd  = (lane >= 16) ? 16 : 0;           // B layout: upper = K+16

    // compute assignment: waves 0-3 -> mtile 0-3 on subtiles 0-3,
    //                     waves 4-7 -> mtile 0-3 on subtiles 4-7
    const int mtile = wave & 3;
    const int nbase = (wave >> 2) * 4;

    const v8f zero = {};
    v8f acc[4];
#pragma unroll
    for (int s = 0; s < 4; ++s) acc[s] = zero;

    const _Float16* aptr =
        Apack + ((size_t)mtile * KSTEPS) * 32 * 16 + (size_t)lane * 16;
    const _Float16* xsg = xs + owLoc;

    __syncthreads();                                   // xs, offB ready

    for (int ks2 = 0; ks2 < KSTEPS; ks2 += 2) {
#pragma unroll
        for (int p = 0; p < 2; ++p) {                  // ping-pong stages
            const int kstep = ks2 + p;

            // gather 16 k-consecutive elements for my column (fragment order)
            const v16us off =
                *reinterpret_cast<const v16us*>(offB + kstep * 32 + kadd);
            v16h tmp;
#pragma unroll
            for (int e = 0; e < 16; ++e)
                tmp[e] = xsg[(int)off[e]];             // ds_load_u16
            Bs[p][tid] = tmp;                          // 2x ds_store_b128
            __syncthreads();   // only barrier this stage: refill of buffer p is
                               // always separated from its previous readers by
                               // the other buffer's barrier

            // A fragment: contiguous, pre-packed, 32B aligned
            const v16h a =
                *reinterpret_cast<const v16h*>(aptr + (size_t)kstep * 32 * 16);

#pragma unroll
            for (int s = 0; s < 4; ++s) {
                const v16h b = Bs[p][(nbase + s) * 32 + lane]; // 2x ds_load_b128
                acc[s] = __builtin_amdgcn_wmma_f32_16x16x32_f16(
                    false, a, false, b, (short)0, acc[s], false, false);
            }
        }
    }

    // epilogue: C layout -> VGPR r holds M=r (lanes 0-15) / M=r+8 (lanes 16-31)
    const int coBase = mtile * 16 + ((lane >= 16) ? 8 : 0);
    const int col    = lane & 15;
#pragma unroll
    for (int s = 0; s < 4; ++s) {
        const int ow = ow0 + (nbase + s) * 16 + col;
        if (ow >= OW) continue;
#pragma unroll
        for (int r = 0; r < 8; ++r) {
            const int co = coBase + r;
            if (co < COUT)
                pots[((((size_t)t * COUT + co) * OH) + oh) * OW + ow] = acc[s][r];
        }
    }
}

// -----------------------------------------------------------------------------
// Decision: first channel-major pooled position with any pots[t=14] >= 15.
// -----------------------------------------------------------------------------
__global__ void init_min(int* minIdx) { *minIdx = INT_MAX; }

__global__ void scan_spikes(const float* __restrict__ pots, int* __restrict__ minIdx) {
    const int idx = blockIdx.x * blockDim.x + threadIdx.x;
    if (idx >= COUT * PH * PW) return;
    const int co  = idx / (PH * PW);
    const int rem = idx - co * (PH * PW);
    const int ph  = rem / PW;
    const int pw  = rem - ph * PW;
    bool spike = false;
#pragma unroll
    for (int dy = 0; dy < 3; ++dy)
#pragma unroll
        for (int dx = 0; dx < 3; ++dx) {
            const int oh = 3 * ph + dy;
            const int ow = 3 * pw + dx;
            const float p = pots[((((size_t)(T_ - 1) * COUT + co) * OH) + oh) * OW + ow];
            spike |= (p >= THRESH);
        }
    if (spike) atomicMin(minIdx, idx);   // min is order-independent: deterministic
}

__global__ void finalize(const int* __restrict__ minIdx, float* __restrict__ out) {
    const int m   = *minIdx;
    const int dec = (m == INT_MAX) ? -1 : ((m / (PH * PW)) / 20);
    out[0] = (float)dec;
}

// -----------------------------------------------------------------------------
extern "C" void kernel_launch(void* const* d_in, const int* in_sizes, int n_in,
                              void* d_out, int out_size, void* d_ws, size_t ws_size,
                              hipStream_t stream) {
    const float* x  = (const float*)d_in[0];   // [15,40,160,300] f32
    const float* W  = (const float*)d_in[1];   // [50,40,4,6]    f32
    float* out      = (float*)d_out;           // [0]=decision, [1..]=pots
    _Float16* Apack = (_Float16*)d_ws;
    int* minIdx     = (int*)((char*)d_ws + MINIDX_OFF);

    pack_A<<<dim3(MTILES * KSTEPS), dim3(32), 0, stream>>>(W, Apack);
    init_min<<<1, 1, 0, stream>>>(minIdx);

    conv_wmma<<<dim3((OW + TCOL - 1) / TCOL, OH, T_), dim3(256), 0, stream>>>(
        x, Apack, out + 1);

    const int nscan = COUT * PH * PW;
    scan_spikes<<<(nscan + 255) / 256, 256, 0, stream>>>(out + 1, minIdx);
    finalize<<<1, 1, 0, stream>>>(minIdx, out);
}
